// MultiScaleRoIAlign_29918742184605
// MI455X (gfx1250) — matrix operand order repaired and checked by reading, not compiled, not run on hardware
//
#include <hip/hip_runtime.h>

// CDNA5 WMMA vector types (per CDNA5_HIP.md, codegen-verified signature)
typedef _Float16 v16h __attribute__((ext_vector_type(16)));
typedef float    v8f  __attribute__((ext_vector_type(8)));

#define C_CH 256
#define POUT 7

// ---------------------------------------------------------------------------
// NCHW -> NHWC transpose (per level). Writes coalesced; reads hit L2 (all
// features fit in the 192MB L2, so the strided reads are cheap).
// ---------------------------------------------------------------------------
__global__ __launch_bounds__(256)
void nchw_to_nhwc(const float* __restrict__ in, float* __restrict__ out,
                  int H, int total) {
    int idx = blockIdx.x * blockDim.x + threadIdx.x;
    if (idx >= total) return;
    int c    = idx & (C_CH - 1);
    int rest = idx >> 8;              // / 256
    int w    = rest % H; rest /= H;   // W == H (square levels)
    int h    = rest % H;
    int b    = rest / H;
    out[idx] = in[(((size_t)b * C_CH + c) * H + h) * (size_t)H + w];
}

// ---------------------------------------------------------------------------
// Multi-scale RoI align, one workgroup (8 wave32) per box.
//   Phase 1: FPN level select + x/y sample setup (torchvision aligned=False).
//   Phase 2: x-interp (+x mask) into LDS  H[pw][c][m]  (m = y-sample*2 + tap).
//   Phase 3: y-contraction out[ph,c] = sum_m WY[ph,m]*H[m,pw,c] as one
//            v_wmma_f32_16x16x32_f16 per (pw, 16-ch tile); WY carries
//            (1-fy)/fy weights, y-validity mask and the 1/4 subsample mean.
//   Phase 4: outT is staged in final [c][bin] order -> division-free,
//            float4-vectorized, fully-coalesced NCHW-order stores.
// ---------------------------------------------------------------------------
__global__ __launch_bounds__(256)
void msroi_wmma_kernel(const float* __restrict__ p0, const float* __restrict__ p1,
                       const float* __restrict__ p2, const float* __restrict__ p3,
                       const float* __restrict__ boxes, float* __restrict__ out,
                       int NB, int channelsLast) {
    __shared__ __align__(64) _Float16 Hbuf[POUT * 64 * 32];  // 28 KB
    __shared__ __align__(16) float outT[64 * 49];            // [c][bin], 12.5 KB
    __shared__ float sfx[14], sfy[14], svx[14], svy[14];
    __shared__ int   sxl[14], sxh[14], syl[14], syh[14];

    const int tid = threadIdx.x;
    const int k   = blockIdx.x;
    const int b   = k / NB;

    // ---- box -> level (FPN assignment) ----
    const float x1 = boxes[4 * k + 0], y1 = boxes[4 * k + 1];
    const float x2 = boxes[4 * k + 2], y2 = boxes[4 * k + 3];
    const float sbox = sqrtf((x2 - x1) * (y2 - y1));
    const float lvf  = floorf(4.0f + log2f(sbox * (1.0f / 224.0f)) + 1e-6f);
    const int   lvl  = (int)fminf(fmaxf(lvf, 2.0f), 5.0f) - 2;

    const int   H     = 200 >> lvl;                    // 200,100,50,25
    const float scale = 0.25f / (float)(1 << lvl);     // exact powers of two
    const float* base = (lvl == 0) ? p0 : (lvl == 1) ? p1 : (lvl == 2) ? p2 : p3;
    base += (size_t)b * C_CH * H * H;                  // same offset NCHW/NHWC
    const int pixStride  = channelsLast ? C_CH : 1;
    const int chanStride = channelsLast ? 1    : H * H;

    const float x1s  = x1 * scale, y1s = y1 * scale;
    const float binw = fmaxf(x2 * scale - x1s, 1.0f) * (1.0f / POUT);
    const float binh = fmaxf(y2 * scale - y1s, 1.0f) * (1.0f / POUT);

    // ---- Phase 1: sample setup (threads 0..13 -> x, 16..29 -> y) ----
    if (tid < 14 || (tid >= 16 && tid < 30)) {
        const bool  isY   = (tid >= 16);
        const int   i     = isY ? tid - 16 : tid;
        const float start = isY ? y1s : x1s;
        const float bin   = isY ? binh : binw;
        // sample position: p + (sub + 0.5)/2, sr==2
        const float v  = start + ((float)(i >> 1) + ((float)(i & 1) + 0.5f) * 0.5f) * bin;
        const float ok = (v >= -1.0f && v <= (float)H) ? 1.0f : 0.0f;
        float vc = fminf(fmaxf(v, 0.0f), (float)(H - 1));
        int lo = (int)floorf(vc);
        lo = (lo < H - 1) ? lo : (H - 1);
        const int   hi2 = ((lo + 1) < (H - 1)) ? (lo + 1) : (H - 1);
        const float fr  = vc - (float)lo;
        if (isY) { syl[i] = lo; syh[i] = hi2; sfy[i] = fr; svy[i] = ok; }
        else     { sxl[i] = lo; sxh[i] = hi2; sfx[i] = fr; svx[i] = ok; }
    }
    __syncthreads();

    // ---- hoist x-sample data into registers (constant-indexed -> VGPRs) ----
    int   offL[14], offH[14];
    float wlo[14], whi[14];
#pragma unroll
    for (int i = 0; i < 14; ++i) {
        offL[i] = sxl[i] * pixStride;
        offH[i] = sxh[i] * pixStride;
        const float fx = sfx[i], vx = svx[i];
        wlo[i] = vx * (1.0f - fx);
        whi[i] = vx * fx;
    }

    // ---- per-lane WMMA A operand (WY weights), this code's contiguous k order (matches B) ----
    // lanes 0-15: M=lane, halves 0..15 -> K=0..15
    // lanes16-31: M=lane-16, halves 0..15 -> K=16..31
    const int lane = tid & 31;
    const int hi   = lane >> 4;
    const int M    = lane & 15;
    v16h aW;
#pragma unroll
    for (int idx = 0; idx < 16; ++idx) {
        const int kk = hi * 16 + idx;
        const int j = kk >> 1, t = kk & 1;   // y-sample j, tap t (lo/hi)
        float w = 0.0f;
        if (M < POUT && j < 14 && (j >> 1) == M)
            w = 0.25f * svy[j] * (t ? sfy[j] : (1.0f - sfy[j]));
        aW[idx] = (_Float16)w;
    }

    const int wave = tid >> 5;
    const int cc   = tid & 63;   // channel within 64-wide chunk (coalesced)
    const int pb   = tid >> 6;   // 0..3: which m rows this group covers

    for (int chunk = 0; chunk < 4; ++chunk) {
        const int c0 = chunk * 64;
        const size_t coff = (size_t)(c0 + cc) * chanStride;

        // ---- Phase 2: x-interpolation into Hbuf[pw][c][m] ----
        for (int mi = 0; mi < 7; ++mi) {
            const int m = pb + 4 * mi;           // 0..27
            const int j = m >> 1, t = m & 1;
            const int row = t ? syh[j] : syl[j];
            const float* rp = base + (size_t)row * H * pixStride + coff;
#pragma unroll
            for (int pw = 0; pw < POUT; ++pw) {
                float acc = 0.0f;
#pragma unroll
                for (int u = 0; u < 2; ++u) {
                    const int i = pw * 2 + u;
                    acc = fmaf(wlo[i], rp[offL[i]], acc);
                    acc = fmaf(whi[i], rp[offH[i]], acc);
                }
                Hbuf[(pw * 64 + cc) * 32 + m] = (_Float16)acc;
            }
        }
        // zero-pad K = 28..31 so the dense WMMA reduction is exact
        for (int i2 = tid; i2 < POUT * 64 * 4; i2 += 256) {
            const int q = i2 & 3, c2 = (i2 >> 2) & 63, pw2 = i2 >> 8;
            Hbuf[(pw2 * 64 + c2) * 32 + 28 + q] = (_Float16)0.0f;
        }
        __syncthreads();

        // ---- Phase 3: one WMMA per (pw, 16-channel tile) ----
        // B, contiguous k order as in A: lanes 0-15 hold K=0..15 (N=lane), lanes16-31 K=16..31
        for (int task = wave; task < 28; task += 8) {
            const int pw = task >> 2;
            const int n0 = (task & 3) * 16;
            const v16h bW = *(const v16h*)&Hbuf[((pw * 64) + (n0 + M)) * 32 + hi * 16];
            v8f d = {};
            d = __builtin_amdgcn_wmma_f32_16x16x32_f16(
                    /*neg_a=*/false, aW, /*neg_b=*/false, bW,
                    /*c_mod=*/(short)0, d, /*reuse_a=*/false, /*reuse_b=*/false);
            if (hi == 0) {   // lanes 0-15 hold D rows M=0..7 (ph 0..6 valid)
#pragma unroll
                for (int v = 0; v < POUT; ++v)
                    outT[(n0 + M) * 49 + v * POUT + pw] = d[v];
            }
        }
        __syncthreads();

        // ---- Phase 4: outT already in final order -> linear float4 copy ----
        {
            const float4* srcv = (const float4*)outT;
            float4* dstv = (float4*)(out + (size_t)k * (C_CH * 49) + (size_t)c0 * 49);
            for (int i4 = tid; i4 < 64 * 49 / 4; i4 += 256)
                dstv[i4] = srcv[i4];
        }
        __syncthreads();
    }
}

// ---------------------------------------------------------------------------
extern "C" void kernel_launch(void* const* d_in, const int* in_sizes, int n_in,
                              void* d_out, int out_size, void* d_ws, size_t ws_size,
                              hipStream_t stream) {
    const float* feats[4] = {(const float*)d_in[0], (const float*)d_in[1],
                             (const float*)d_in[2], (const float*)d_in[3]};
    const float* boxes = (const float*)d_in[5];
    const int HS[4] = {200, 100, 50, 25};

    int B = in_sizes[0] / (C_CH * 200 * 200);
    if (B < 1) B = 1;
    const int K  = in_sizes[5] / 4;
    const int NB = K / B;

    size_t off[4]; size_t tot = 0;
    for (int l = 0; l < 4; ++l) { off[l] = tot; tot += (size_t)B * C_CH * HS[l] * HS[l]; }
    const bool cl = (ws_size >= tot * sizeof(float));

    const float* base[4];
    if (cl) {
        float* ws = (float*)d_ws;
        for (int l = 0; l < 4; ++l) {
            const int total = B * C_CH * HS[l] * HS[l];
            const int grid  = (total + 255) / 256;
            nchw_to_nhwc<<<grid, 256, 0, stream>>>(feats[l], ws + off[l], HS[l], total);
            base[l] = ws + off[l];
        }
    } else {
        for (int l = 0; l < 4; ++l) base[l] = feats[l];
    }

    msroi_wmma_kernel<<<K, 256, 0, stream>>>(base[0], base[1], base[2], base[3],
                                             boxes, (float*)d_out, NB, cl ? 1 : 0);
    (void)n_in; (void)out_size;
}
